// QRNN_56521769615747
// MI455X (gfx1250) — matrix operand order repaired and checked by reading, not compiled, not run on hardware
//
#include <hip/hip_runtime.h>

// ---------- CDNA5 types ----------
typedef __attribute__((ext_vector_type(16))) __bf16         v16bf;
typedef __attribute__((ext_vector_type(8)))  float          v8f;
typedef __attribute__((ext_vector_type(16))) unsigned short v16u;
typedef __attribute__((ext_vector_type(8)))  unsigned short v8u;
typedef __attribute__((ext_vector_type(4)))  unsigned short v4u;

#define B_   8
#define T_   2048
#define C_   512
#define U_   512
#define NG   1536           // 3*U
#define BT   (B_ * T_)      // 16384
#define SCH  16             // scan chunks per sequence
#define LCH  (T_ / SCH)     // 128 steps per chunk

#define LDS_STRIDE 40
#define ABUF (128 * LDS_STRIDE)      // ushorts per tile buffer
#define BBASE (2 * ABUF)             // Bs buffers start after 2 A buffers

__device__ __forceinline__ unsigned short f2bf(float x) {
  unsigned int u = __float_as_uint(x);
  u += 0x7fffu + ((u >> 16) & 1u);   // round-to-nearest-even
  return (unsigned short)(u >> 16);
}

__device__ __forceinline__ void wait_async_le4() {
#if __has_builtin(__builtin_amdgcn_s_wait_asynccnt)
  __builtin_amdgcn_s_wait_asynccnt(4);
#else
  asm volatile("s_wait_asynccnt 0x4" ::: "memory");
#endif
}
__device__ __forceinline__ void wait_async_0() {
#if __has_builtin(__builtin_amdgcn_s_wait_asynccnt)
  __builtin_amdgcn_s_wait_asynccnt(0);
#else
  asm volatile("s_wait_asynccnt 0x0" ::: "memory");
#endif
}

// ---------- kernel 1: x fp32 -> bf16 (+ zero row init) ----------
__global__ __launch_bounds__(256) void qrnn_cvt_x(const float* __restrict__ x,
                                                  unsigned short* __restrict__ xb,
                                                  unsigned short* __restrict__ zrow) {
  const int i = (blockIdx.x * 256 + threadIdx.x) * 4;
  const float4 v = *(const float4*)(x + i);
  v4u o;
  o[0] = f2bf(v.x); o[1] = f2bf(v.y); o[2] = f2bf(v.z); o[3] = f2bf(v.w);
  *(v4u*)(xb + i) = o;
  if (blockIdx.x == 0) {             // 512 ushorts of zeros for the causal pad
    zrow[threadIdx.x * 2]     = 0;
    zrow[threadIdx.x * 2 + 1] = 0;
  }
}

// ---------- kernel 2: kernel fp32 [W,C,3U] -> bf16 transposed [W,3U,C] ----------
__global__ __launch_bounds__(256) void qrnn_cvt_k(const float* __restrict__ k,
                                                  unsigned short* __restrict__ kbT) {
  const int i = blockIdx.x * 256 + threadIdx.x;   // over W*NG*C
  const int c = i & (C_ - 1);
  const int n = (i >> 9) % NG;
  const int w = i / (NG * C_);
  kbT[i] = f2bf(k[(w * C_ + c) * NG + n]);
}

// Issue async global->LDS staging for stage s (4 x b128 per thread = 64B)
#define ISSUE_STAGE(s)                                                         \
  do {                                                                         \
    const unsigned _buf = (unsigned)((s) & 1);                                 \
    const unsigned short* _sa = ((s) < 16) ? (aSrc0 + (s) * 32)                \
                                           : (aSrc1 + ((s) - 16) * 32);        \
    const unsigned short* _sb = ((s) < 16) ? (bSrc0 + (s) * 32)                \
                                           : (bSrc1 + ((s) - 16) * 32);        \
    unsigned _la = ldsBase + _buf * (ABUF * 2) + sOff;                         \
    unsigned _lb = ldsBase + (BBASE + _buf * ABUF) * 2 + sOff;                 \
    unsigned long long _ga = (unsigned long long)(size_t)_sa;                  \
    unsigned long long _gb = (unsigned long long)(size_t)_sb;                  \
    asm volatile("global_load_async_to_lds_b128 %0, %1, off"                   \
                 :: "v"(_la), "v"(_ga) : "memory");                            \
    asm volatile("global_load_async_to_lds_b128 %0, %1, off offset:16"         \
                 :: "v"(_la), "v"(_ga) : "memory");                            \
    asm volatile("global_load_async_to_lds_b128 %0, %1, off"                   \
                 :: "v"(_lb), "v"(_gb) : "memory");                            \
    asm volatile("global_load_async_to_lds_b128 %0, %1, off offset:16"         \
                 :: "v"(_lb), "v"(_gb) : "memory");                            \
  } while (0)

// ---------- kernel 3: fused conv-GEMM (bf16 WMMA, async double-buffered LDS) ----------
// grid: (NG/128, BT/128), block: 256 = 8 waves (4M x 2N); wave tile 32x64 = 2x4 WMMAs.
__global__ __launch_bounds__(256) void qrnn_gemm(
    const unsigned short* __restrict__ xb,
    const unsigned short* __restrict__ kbT,
    const unsigned short* __restrict__ zrow,
    const float* __restrict__ bias,
    float* __restrict__ Z, float* __restrict__ F, float* __restrict__ O) {
  __shared__ unsigned short smem[4 * ABUF];   // As[2] | Bs[2], 40960 B

  const int n0   = blockIdx.x * 128;
  const int m0   = blockIdx.y * 128;
  const int tid  = threadIdx.x;
  const int lane = tid & 31;
  const int wave = tid >> 5;
  const int wm   = wave & 3;
  const int wn   = wave >> 2;
  const int lrow = lane & 15;
  const int lhi  = lane >> 4;
  const int r    = tid >> 1;     // staging row 0..127
  const int half = tid & 1;

  // Branch-free staging sources (causal pad rows read the zero row instead)
  const int m = m0 + r;
  const int t = m & (T_ - 1);
  const unsigned short* aSrc0 = (t == 0) ? (zrow + half * 16)
                                         : (xb + (size_t)(m - 1) * C_ + half * 16);
  const unsigned short* aSrc1 = xb + (size_t)m * C_ + half * 16;
  const unsigned short* bSrc0 = kbT + (size_t)(n0 + r) * C_ + half * 16;
  const unsigned short* bSrc1 = bSrc0 + (size_t)NG * C_;

  // LDS byte addresses (generic LDS pointer low 32 bits == LDS offset)
  const unsigned ldsBase = (unsigned)(size_t)(&smem[0]);
  const unsigned sOff    = (unsigned)((r * LDS_STRIDE + half * 16) * 2);

  v8f acc[2][4] = {};

  ISSUE_STAGE(0);
  ISSUE_STAGE(1);

#pragma unroll
  for (int kt = 0; kt < 32; ++kt) {
    if (kt < 31) wait_async_le4();   // stage kt landed (in-order completion)
    else         wait_async_0();
    __syncthreads();

    const unsigned short* As = &smem[(kt & 1) * ABUF];
    const unsigned short* Bs = &smem[BBASE + (kt & 1) * ABUF];

    v16bf afrag[2], bfrag[4];
#pragma unroll
    for (int fi = 0; fi < 2; ++fi) {
      const int row = wm * 32 + fi * 16 + lrow;
      v8u cA = *(const v8u*)&As[row * LDS_STRIDE + lhi * 8];
      v8u cB = *(const v8u*)&As[row * LDS_STRIDE + lhi * 8 + 16];
      v16u a = __builtin_shufflevector(cA, cB, 0,1,2,3,4,5,6,7,8,9,10,11,12,13,14,15);
      afrag[fi] = __builtin_bit_cast(v16bf, a);
    }
#pragma unroll
    for (int fj = 0; fj < 4; ++fj) {
      const int n = wn * 64 + fj * 16 + lrow;
      v8u cA = *(const v8u*)&Bs[n * LDS_STRIDE + lhi * 8];
      v8u cB = *(const v8u*)&Bs[n * LDS_STRIDE + lhi * 8 + 16];
      v16u b = __builtin_shufflevector(cA, cB, 0,1,2,3,4,5,6,7,8,9,10,11,12,13,14,15);
      bfrag[fj] = __builtin_bit_cast(v16bf, b);
    }

#pragma unroll
    for (int fi = 0; fi < 2; ++fi)
#pragma unroll
      for (int fj = 0; fj < 4; ++fj)
        acc[fi][fj] = __builtin_amdgcn_wmma_f32_16x16x32_bf16(
            false, afrag[fi], false, bfrag[fj], (short)0, acc[fi][fj], false, false);

    __syncthreads();                 // everyone done reading buf (kt&1)
    if (kt + 2 < 32) {
      ISSUE_STAGE(kt + 2);           // refill the buffer just consumed
    }
  }

  // ---- fused epilogue: each 128-wide N tile is purely one gate type ----
  const int g = n0 >> 9;             // 0=z, 1=f, 2=o
  float* outp = (g == 0) ? Z : ((g == 1) ? F : O);
#pragma unroll
  for (int fi = 0; fi < 2; ++fi) {
#pragma unroll
    for (int fj = 0; fj < 4; ++fj) {
      const int col = n0 + wn * 64 + fj * 16 + lrow;
      const int u   = col - g * 512;
      const float bia = bias[col];
#pragma unroll
      for (int v = 0; v < 8; ++v) {
        const int mrow = m0 + wm * 32 + fi * 16 + lhi * 8 + v;  // C/D layout
        const float gv = acc[fi][fj][v] + bia;
        const float a  = (g == 0) ? tanhf(gv) : (1.0f / (1.0f + __expf(-gv)));
        outp[(size_t)mrow * U_ + u] = a;
      }
    }
  }
}

// ---------- kernel 4a: per-chunk affine composition (A = prod f, B = fold) ----------
__global__ __launch_bounds__(256) void qrnn_scan_part(
    const float* __restrict__ Z, const float* __restrict__ F,
    float* __restrict__ Ach, float* __restrict__ Bch) {
  const int idx = blockIdx.x * 256 + threadIdx.x;   // 0 .. B*SCH*U-1
  const int u = idx & (U_ - 1);
  const int s = (idx >> 9) & (SCH - 1);
  const int b = idx >> 13;
  const size_t base = ((size_t)b * T_ + (size_t)s * LCH) * U_ + u;
  float A = 1.0f, Bc = 0.0f;
#pragma unroll 4
  for (int tt = 0; tt < LCH; ++tt) {
    const size_t i = base + (size_t)tt * U_;
    const float f = F[i];
    const float z = Z[i];
    A  = f * A;
    Bc = f * (Bc - z) + z;           // f*Bc + (1-f)*z
  }
  Ach[idx] = A;
  Bch[idx] = Bc;
}

// ---------- kernel 4b: scan the chunk states (16 per channel) ----------
__global__ __launch_bounds__(256) void qrnn_scan_mid(
    const float* __restrict__ Ach, const float* __restrict__ Bch,
    const float* __restrict__ h0, float* __restrict__ Hin) {
  const int ch = blockIdx.x * 256 + threadIdx.x;    // 0..B*U-1
  const int u = ch & (U_ - 1);
  const int b = ch >> 9;
  float h = h0[u];
#pragma unroll
  for (int s = 0; s < SCH; ++s) {
    const int idx = ((b * SCH) + s) * U_ + u;
    Hin[idx] = h;                    // state entering chunk s
    h = Ach[idx] * h + Bch[idx];
  }
}

// ---------- kernel 4c: replay chunks from corrected entry state + output gate ----------
__global__ __launch_bounds__(256) void qrnn_scan_final(
    const float* __restrict__ Z, const float* __restrict__ F,
    const float* __restrict__ Hin, float* __restrict__ out) {
  const int idx = blockIdx.x * 256 + threadIdx.x;
  const int u = idx & (U_ - 1);
  const int s = (idx >> 9) & (SCH - 1);
  const int b = idx >> 13;
  const size_t base = ((size_t)b * T_ + (size_t)s * LCH) * U_ + u;
  float h = Hin[idx];
#pragma unroll 4
  for (int tt = 0; tt < LCH; ++tt) {
    const size_t i = base + (size_t)tt * U_;
    const float f = F[i];
    const float z = Z[i];
    const float o = out[i];          // o written by GEMM epilogue
    h = f * (h - z) + z;
    out[i] = h * o;
  }
}

extern "C" void kernel_launch(void* const* d_in, const int* in_sizes, int n_in,
                              void* d_out, int out_size, void* d_ws, size_t ws_size,
                              hipStream_t stream) {
  const float* x    = (const float*)d_in[0];
  const float* kern = (const float*)d_in[1];
  const float* bias = (const float*)d_in[2];
  const float* h0   = (const float*)d_in[3];
  float* out = (float*)d_out;

  char* ws = (char*)d_ws;
  size_t off = 0;
  unsigned short* xb  = (unsigned short*)(ws + off); off += (size_t)BT * C_ * 2;
  unsigned short* kbT = (unsigned short*)(ws + off); off += (size_t)2 * NG * C_ * 2;
  float* Z   = (float*)(ws + off); off += (size_t)BT * U_ * 4;
  float* F   = (float*)(ws + off); off += (size_t)BT * U_ * 4;
  float* Ach = (float*)(ws + off); off += (size_t)B_ * SCH * U_ * 4;
  float* Bch = (float*)(ws + off); off += (size_t)B_ * SCH * U_ * 4;
  float* Hin = (float*)(ws + off); off += (size_t)B_ * SCH * U_ * 4;
  unsigned short* zrow = (unsigned short*)(ws + off); off += 1024;

  qrnn_cvt_x<<<(BT * C_) / 1024, 256, 0, stream>>>(x, xb, zrow);
  qrnn_cvt_k<<<(2 * NG * C_) / 256, 256, 0, stream>>>(kern, kbT);
  qrnn_gemm<<<dim3(NG / 128, BT / 128), 256, 0, stream>>>(xb, kbT, zrow, bias, Z, F, out);
  qrnn_scan_part<<<(B_ * SCH * U_) / 256, 256, 0, stream>>>(Z, F, Ach, Bch);
  qrnn_scan_mid<<<(B_ * U_) / 256, 256, 0, stream>>>(Ach, Bch, h0, Hin);
  qrnn_scan_final<<<(B_ * SCH * U_) / 256, 256, 0, stream>>>(Z, F, Hin, out);
}
